// GATRoutePredictor_74131135529944
// MI455X (gfx1250) — matrix operand order repaired and checked by reading, not compiled, not run on hardware
//
#include <hip/hip_runtime.h>
#include <hip/hip_bf16.h>
#include <math.h>

typedef __bf16 bf16_t;
typedef __attribute__((ext_vector_type(16))) __bf16 v16bf;
typedef __attribute__((ext_vector_type(8)))  __bf16 v8bf;
typedef __attribute__((ext_vector_type(8)))  float  v8f;

static constexpr int NN    = 50000;
static constexpr int NE    = 800000;
static constexpr int ETOT  = NE + NN;     // edges + self loops
static constexpr int NH    = 4;           // heads (layers 0/1)
static constexpr int HC    = 64;          // per-head channels
static constexpr int DIM   = 256;         // NH*HC
static constexpr float SLOPE = 0.2f;
static constexpr float EPS   = 1e-5f;

// ---------------- conversion kernels ----------------
__global__ void k_cvt_bf16(const float* __restrict__ x, bf16_t* __restrict__ y, int n) {
  int i = blockIdx.x * blockDim.x + threadIdx.x;
  if (i < n) y[i] = (bf16_t)x[i];
}

// W [K,M] f32 row-major -> Wt [M,K] bf16 row-major
__global__ void k_cvt_wt(const float* __restrict__ W, bf16_t* __restrict__ Wt, int K, int M) {
  int i = blockIdx.x * blockDim.x + threadIdx.x;
  if (i >= K * M) return;
  int k = i / M, m = i % M;
  Wt[(size_t)m * K + k] = (bf16_t)W[i];
}

__global__ void k_fill(float* __restrict__ p, int n, float v) {
  int i = blockIdx.x * blockDim.x + threadIdx.x;
  if (i < n) p[i] = v;
}

__global__ void k_init_bias(float* __restrict__ o, const float* __restrict__ b, int n) {
  int i = blockIdx.x * blockDim.x + threadIdx.x;
  if (i < n) o[i] = b[i & (DIM - 1)];
}

__global__ void k_init_scalar(float* __restrict__ o, const float* __restrict__ b, int n) {
  int i = blockIdx.x * blockDim.x + threadIdx.x;
  if (i < n) o[i] = b[0];
}

// ---------------- helpers for WMMA fragments ----------------
__device__ inline v16bf frag_a(const bf16_t* __restrict__ arow, int kk, int hi) {
  // ISA 16-bit A 16x32 layout: lane half picks K sub-chunks [hi*8, hi*8+8) and +16
  v8bf lo = *(const v8bf*)(arow + kk + hi * 8);
  v8bf h8 = *(const v8bf*)(arow + kk + 16 + hi * 8);
  v16bf a;
#pragma unroll
  for (int i = 0; i < 8; ++i) { a[i] = lo[i]; a[i + 8] = h8[i]; }
  return a;
}

__device__ inline v16bf frag_b(const bf16_t* __restrict__ brow, int kk, int hi) {
  // B 32x16 from pre-transposed weights: lane half picks K sub-chunks [hi*8, hi*8+8) and +16
  v8bf lo = *(const v8bf*)(brow + kk + hi * 8);
  v8bf h8 = *(const v8bf*)(brow + kk + 16 + hi * 8);
  v16bf b;
#pragma unroll
  for (int i = 0; i < 8; ++i) { b[i] = lo[i]; b[i + 8] = h8[i]; }
  return b;
}

// ---------------- WMMA GEMM: C[N,M] = A[N,K](bf16) * Bt[M,K](bf16)^T ----------------
// one wave per 16x64 output strip: A fragment loaded once, fed to 4 WMMAs.
// 4 waves / 128-thread block.
__global__ __launch_bounds__(128)
void k_gemm_wmma4(const bf16_t* __restrict__ A, const bf16_t* __restrict__ Bt,
                  float* __restrict__ C, int K, int M, int numWaves) {
  int wave = (blockIdx.x * blockDim.x + threadIdx.x) >> 5;
  if (wave >= numWaves) return;                    // wave-uniform branch
  int cg = M >> 6;                                 // column groups of 64
  int rowTile = wave / cg;
  int colBase = (wave - rowTile * cg) * 64;
  int lane = threadIdx.x & 31;
  int hi   = lane >> 4;
  int mr   = lane & 15;
  const bf16_t* arow = A  + (size_t)(rowTile * 16 + mr) * K;
  const bf16_t* br0  = Bt + (size_t)(colBase +  0 + mr) * K;
  const bf16_t* br1  = Bt + (size_t)(colBase + 16 + mr) * K;
  const bf16_t* br2  = Bt + (size_t)(colBase + 32 + mr) * K;
  const bf16_t* br3  = Bt + (size_t)(colBase + 48 + mr) * K;
  v8f acc0 = {}, acc1 = {}, acc2 = {}, acc3 = {};
  for (int kk = 0; kk < K; kk += 32) {
    v16bf a  = frag_a(arow, kk, hi);
    v16bf b0 = frag_b(br0, kk, hi);
    v16bf b1 = frag_b(br1, kk, hi);
    v16bf b2 = frag_b(br2, kk, hi);
    v16bf b3 = frag_b(br3, kk, hi);
    acc0 = __builtin_amdgcn_wmma_f32_16x16x32_bf16(false, a, false, b0, (short)0, acc0, false, false);
    acc1 = __builtin_amdgcn_wmma_f32_16x16x32_bf16(false, a, false, b1, (short)0, acc1, false, false);
    acc2 = __builtin_amdgcn_wmma_f32_16x16x32_bf16(false, a, false, b2, (short)0, acc2, false, false);
    acc3 = __builtin_amdgcn_wmma_f32_16x16x32_bf16(false, a, false, b3, (short)0, acc3, false, false);
  }
  // C/D layout: lane holds rows v + hi*8, column = colTile*16 + mr
  float* cbase = C + (size_t)(rowTile * 16 + hi * 8) * M + colBase + mr;
#pragma unroll
  for (int v = 0; v < 8; ++v) {
    float* cr = cbase + (size_t)v * M;
    cr[0]  = acc0[v];
    cr[16] = acc1[v];
    cr[32] = acc2[v];
    cr[48] = acc3[v];
  }
}

// ---------------- attention node scores ----------------
__global__ void k_scores(const float* __restrict__ H, const float* __restrict__ a_s,
                         const float* __restrict__ a_d, float* __restrict__ s,
                         float* __restrict__ t, int n) {
  int i = blockIdx.x * blockDim.x + threadIdx.x;
  if (i >= n * NH) return;
  int node = i / NH, h = i - node * NH;
  const float* hr = H + (size_t)node * DIM + h * HC;
  float ss = 0.f, tt = 0.f;
#pragma unroll 4
  for (int c = 0; c < HC; ++c) { ss += hr[c] * a_s[h * HC + c]; tt += hr[c] * a_d[h * HC + c]; }
  s[i] = ss; t[i] = tt;
}

// ---------------- edge kernels ----------------
__device__ inline void edge_ep(int e, const int* __restrict__ ei, int& s, int& d) {
  if (e < NE) { s = ei[e]; d = ei[NE + e]; }
  else        { s = d = e - NE; }
}

__device__ inline void atomicMaxF(float* addr, float v) {
  if (v >= 0.f) atomicMax((int*)addr, __float_as_int(v));
  else          atomicMin((unsigned int*)addr, __float_as_uint(v));
}

__global__ void k_edge_max(const int* __restrict__ ei, const float* __restrict__ s,
                           const float* __restrict__ t, float* __restrict__ m, int H) {
  int i = blockIdx.x * blockDim.x + threadIdx.x;
  if (i >= ETOT * H) return;
  int e = i / H, h = i - e * H, sn, dn;
  edge_ep(e, ei, sn, dn);
  float sc = s[sn * H + h] + t[dn * H + h];
  sc = sc > 0.f ? sc : SLOPE * sc;
  atomicMaxF(&m[dn * H + h], sc);
}

__global__ void k_edge_expsum(const int* __restrict__ ei, const float* __restrict__ s,
                              const float* __restrict__ t, const float* __restrict__ m,
                              float* __restrict__ den, float* __restrict__ ex, int H) {
  int i = blockIdx.x * blockDim.x + threadIdx.x;
  if (i >= ETOT * H) return;
  int e = i / H, h = i - e * H, sn, dn;
  edge_ep(e, ei, sn, dn);
  float sc = s[sn * H + h] + t[dn * H + h];
  sc = sc > 0.f ? sc : SLOPE * sc;
  float v = expf(sc - m[dn * H + h]);
  ex[i] = v;
  atomicAdd(&den[dn * H + h], v);
}

// message scatter for 4-head, 64-channel layers; thread = (edge, head, 4-chan group)
__global__ void k_edge_msg4(const int* __restrict__ ei, const float* __restrict__ ex,
                            const float* __restrict__ den, const float* __restrict__ H,
                            float* __restrict__ out) {
  long i = (long)blockIdx.x * blockDim.x + threadIdx.x;
  if (i >= (long)ETOT * NH * 16) return;
  int c4 = (int)(i & 15);
  int h  = (int)((i >> 4) & 3);
  int e  = (int)(i >> 6);
  int sn, dn; edge_ep(e, ei, sn, dn);
  float alpha = ex[e * NH + h] / den[dn * NH + h];
  const float4 hv = *(const float4*)(H + (size_t)sn * DIM + h * HC + c4 * 4);
  float* o = out + (size_t)dn * DIM + h * HC + c4 * 4;
  atomicAdd(o + 0, hv.x * alpha);
  atomicAdd(o + 1, hv.y * alpha);
  atomicAdd(o + 2, hv.z * alpha);
  atomicAdd(o + 3, hv.w * alpha);
}

// scalar message scatter for layer 2 (H=1, C=1)
__global__ void k_edge_msg1(const int* __restrict__ ei, const float* __restrict__ ex,
                            const float* __restrict__ den, const float* __restrict__ h2,
                            float* __restrict__ out) {
  int e = blockIdx.x * blockDim.x + threadIdx.x;
  if (e >= ETOT) return;
  int sn, dn; edge_ep(e, ei, sn, dn);
  atomicAdd(&out[dn], h2[sn] * (ex[e] / den[dn]));
}

// ---------------- LayerNorm + ELU (wave32 per node), emits bf16 for next GEMM ----------------
__global__ __launch_bounds__(256)
void k_ln_elu(const float* __restrict__ X, const float* __restrict__ g,
              const float* __restrict__ b, bf16_t* __restrict__ Y, int n) {
  int wave = (blockIdx.x * blockDim.x + threadIdx.x) >> 5;
  int lane = threadIdx.x & 31;
  if (wave >= n) return;
  const float* xr = X + (size_t)wave * DIM;
  float v[8];
  float s = 0.f;
#pragma unroll
  for (int j = 0; j < 8; ++j) { v[j] = xr[j * 32 + lane]; s += v[j]; }
#pragma unroll
  for (int o = 16; o > 0; o >>= 1) s += __shfl_xor(s, o, 32);
  float mu = s * (1.f / DIM);
  float vs = 0.f;
#pragma unroll
  for (int j = 0; j < 8; ++j) { float d = v[j] - mu; vs += d * d; }
#pragma unroll
  for (int o = 16; o > 0; o >>= 1) vs += __shfl_xor(vs, o, 32);
  float inv = rsqrtf(vs * (1.f / DIM) + EPS);
#pragma unroll
  for (int j = 0; j < 8; ++j) {
    int f = j * 32 + lane;
    float y = (v[j] - mu) * inv * g[f] + b[f];
    y = y > 0.f ? y : expf(y) - 1.f;
    Y[(size_t)wave * DIM + f] = (bf16_t)y;
  }
}

// ---------------- thin layer-2 projection (wave32 per node) ----------------
__global__ __launch_bounds__(256)
void k_dot2(const bf16_t* __restrict__ A, const float* __restrict__ W2,
            const float* __restrict__ as2, const float* __restrict__ ad2,
            float* __restrict__ h2, float* __restrict__ s2, float* __restrict__ t2, int n) {
  int wave = (blockIdx.x * blockDim.x + threadIdx.x) >> 5;
  int lane = threadIdx.x & 31;
  if (wave >= n) return;
  const bf16_t* ar = A + (size_t)wave * DIM;
  float acc = 0.f;
#pragma unroll
  for (int j = 0; j < 8; ++j) { int k = j * 32 + lane; acc += (float)ar[k] * W2[k]; }
#pragma unroll
  for (int o = 16; o > 0; o >>= 1) acc += __shfl_xor(acc, o, 32);
  if (lane == 0) {
    h2[wave] = acc;
    s2[wave] = acc * as2[0];
    t2[wave] = acc * ad2[0];
  }
}

// ---------------- launcher ----------------
extern "C" void kernel_launch(void* const* d_in, const int* in_sizes, int n_in,
                              void* d_out, int out_size, void* d_ws, size_t ws_size,
                              hipStream_t stream) {
  const float* x    = (const float*)d_in[0];
  const int*   ei   = (const int*)d_in[1];
  const float* W0   = (const float*)d_in[2];
  const float* as0  = (const float*)d_in[3];
  const float* ad0  = (const float*)d_in[4];
  const float* b0   = (const float*)d_in[5];
  const float* W1   = (const float*)d_in[6];
  const float* as1  = (const float*)d_in[7];
  const float* ad1  = (const float*)d_in[8];
  const float* b1   = (const float*)d_in[9];
  const float* W2   = (const float*)d_in[10];
  const float* as2  = (const float*)d_in[11];
  const float* ad2  = (const float*)d_in[12];
  const float* b2   = (const float*)d_in[13];
  const float* ln_g = (const float*)d_in[14];
  const float* ln_b = (const float*)d_in[15];
  float* out = (float*)d_out;

  char* wsp = (char*)d_ws;
  size_t off = 0;
  auto take = [&](size_t bytes) -> void* {
    void* p = wsp + off;
    off = (off + bytes + 255) & ~(size_t)255;
    return p;
  };
  bf16_t* Abf = (bf16_t*)take((size_t)NN * DIM * 2);   // bf16 activations (also holds x bf16)
  float*  Hb  = (float*)take((size_t)NN * DIM * 4);    // GEMM output h
  float*  Ob  = (float*)take((size_t)NN * DIM * 4);    // aggregated output
  bf16_t* W0t = (bf16_t*)take((size_t)DIM * 64 * 2);
  bf16_t* W1t = (bf16_t*)take((size_t)DIM * DIM * 2);
  float*  sb  = (float*)take((size_t)NN * NH * 4);
  float*  tb  = (float*)take((size_t)NN * NH * 4);
  float*  mb  = (float*)take((size_t)NN * NH * 4);
  float*  db  = (float*)take((size_t)NN * NH * 4);
  float*  exb = (float*)take((size_t)ETOT * NH * 4);
  float*  h2  = (float*)take((size_t)NN * 4);

  auto blks = [](long n, int t) { return (unsigned)((n + t - 1) / t); };
  const int T = 256;
  const int gemmWaves  = (NN / 16) * (DIM / 64);       // 3125 * 4 = 12500 strips
  const unsigned gemmBlocks = gemmWaves / 4;           // 4 waves / 128-thread block, exact
  const unsigned waveBlocks = blks((long)NN * 32, T);  // wave-per-node kernels

  // ---- layer 0 ----
  k_cvt_bf16<<<blks((long)NN * 64, T), T, 0, stream>>>(x, Abf, NN * 64);
  k_cvt_wt<<<blks(64L * DIM, T), T, 0, stream>>>(W0, W0t, 64, DIM);
  k_cvt_wt<<<blks((long)DIM * DIM, T), T, 0, stream>>>(W1, W1t, DIM, DIM);
  k_gemm_wmma4<<<gemmBlocks, 128, 0, stream>>>(Abf, W0t, Hb, 64, DIM, gemmWaves);
  k_scores<<<blks((long)NN * NH, T), T, 0, stream>>>(Hb, as0, ad0, sb, tb, NN);
  k_fill<<<blks((long)NN * NH, T), T, 0, stream>>>(mb, NN * NH, -INFINITY);
  k_fill<<<blks((long)NN * NH, T), T, 0, stream>>>(db, NN * NH, 0.f);
  k_init_bias<<<blks((long)NN * DIM, T), T, 0, stream>>>(Ob, b0, NN * DIM);
  k_edge_max<<<blks((long)ETOT * NH, T), T, 0, stream>>>(ei, sb, tb, mb, NH);
  k_edge_expsum<<<blks((long)ETOT * NH, T), T, 0, stream>>>(ei, sb, tb, mb, db, exb, NH);
  k_edge_msg4<<<blks((long)ETOT * NH * 16, T), T, 0, stream>>>(ei, exb, db, Hb, Ob);
  k_ln_elu<<<waveBlocks, T, 0, stream>>>(Ob, ln_g, ln_b, Abf, NN);

  // ---- layer 1 ----
  k_gemm_wmma4<<<gemmBlocks, 128, 0, stream>>>(Abf, W1t, Hb, DIM, DIM, gemmWaves);
  k_scores<<<blks((long)NN * NH, T), T, 0, stream>>>(Hb, as1, ad1, sb, tb, NN);
  k_fill<<<blks((long)NN * NH, T), T, 0, stream>>>(mb, NN * NH, -INFINITY);
  k_fill<<<blks((long)NN * NH, T), T, 0, stream>>>(db, NN * NH, 0.f);
  k_init_bias<<<blks((long)NN * DIM, T), T, 0, stream>>>(Ob, b1, NN * DIM);
  k_edge_max<<<blks((long)ETOT * NH, T), T, 0, stream>>>(ei, sb, tb, mb, NH);
  k_edge_expsum<<<blks((long)ETOT * NH, T), T, 0, stream>>>(ei, sb, tb, mb, db, exb, NH);
  k_edge_msg4<<<blks((long)ETOT * NH * 16, T), T, 0, stream>>>(ei, exb, db, Hb, Ob);
  k_ln_elu<<<waveBlocks, T, 0, stream>>>(Ob, ln_g, ln_b, Abf, NN);

  // ---- layer 2 (H=1, C=1) ----
  k_dot2<<<waveBlocks, T, 0, stream>>>(Abf, W2, as2, ad2, h2, sb, tb, NN);
  k_fill<<<blks(NN, T), T, 0, stream>>>(mb, NN, -INFINITY);
  k_fill<<<blks(NN, T), T, 0, stream>>>(db, NN, 0.f);
  k_init_scalar<<<blks(NN, T), T, 0, stream>>>(out, b2, NN);
  k_edge_max<<<blks(ETOT, T), T, 0, stream>>>(ei, sb, tb, mb, 1);
  k_edge_expsum<<<blks(ETOT, T), T, 0, stream>>>(ei, sb, tb, mb, db, exb, 1);
  k_edge_msg1<<<blks(ETOT, T), T, 0, stream>>>(ei, exb, db, h2, out);
}
